// DeltaNet_19310172962952
// MI455X (gfx1250) — matrix operand order repaired and checked by reading, compiled, not run on hardware
//
#include <hip/hip_runtime.h>
#include <hip/hip_bf16.h>

// Problem constants (match the reference)
#define BB      2
#define LL      4096
#define HIDD    2048
#define HH      16
#define DD      128
#define KWW     4
#define CC      64
#define NCHUNK  64
#define EPSF    1e-5f

typedef __attribute__((ext_vector_type(16))) __bf16 bf16x16;
typedef __attribute__((ext_vector_type(8)))  __bf16 bf16x8;
typedef __attribute__((ext_vector_type(8)))  float  f32x8;
typedef __attribute__((ext_vector_type(4)))  int    i32x4;

// ---------------------------------------------------------------------------
// CDNA5 async global->LDS path (signature probed: (v4i32 AS1*, v4i32 AS3*,
// imm offset, imm cpol); tracked by ASYNCcnt)
// ---------------------------------------------------------------------------
#if defined(__has_builtin)
#if __has_builtin(__builtin_amdgcn_global_load_async_to_lds_b128)
#define HAVE_ASYNC_LDS 1
#endif
#if __has_builtin(__builtin_amdgcn_s_wait_asynccnt)
#define HAVE_WAIT_ASYNC 1
#endif
#endif

#ifdef HAVE_ASYNC_LDS
static __device__ __forceinline__ void async_copy16(const __bf16* g, __bf16* l) {
  __builtin_amdgcn_global_load_async_to_lds_b128(
      (__attribute__((address_space(1))) i32x4*)g,
      (__attribute__((address_space(3))) i32x4*)l, 0, 0);
}
static __device__ __forceinline__ void wait_async0() {
#ifdef HAVE_WAIT_ASYNC
  __builtin_amdgcn_s_wait_asynccnt(0);
#else
  asm volatile("s_wait_asynccnt 0" ::: "memory");
#endif
}
#endif

// ---------------------------------------------------------------------------
// WMMA fragment helpers (v_wmma_f32_16x16x32_bf16, wave32)
// A 16x32 (MxK): lane<16 -> M=lane, K halves [0..7] & [16..23]; lane>=16 -> +8
// B 32x16 as [N][K] in LDS: lane<16 -> N=lane, K halves [0..7] & [16..23]; lane>=16 -> +8
// C/D 16x16 f32: vgpr g, lane l -> M = g + 8*(l>>4), N = l&15
// ---------------------------------------------------------------------------
static __device__ __forceinline__ bf16x16 frag_a(const __bf16* base, int row,
                                                 int stride, int lane) {
  const __bf16* p = base + (size_t)(row + (lane & 15)) * stride + ((lane & 16) ? 8 : 0);
  bf16x8 lo = *(const bf16x8*)(p);
  bf16x8 hi = *(const bf16x8*)(p + 16);
  bf16x16 a;
#pragma unroll
  for (int i = 0; i < 8; ++i) { a[i] = lo[i]; a[i + 8] = hi[i]; }
  return a;
}

static __device__ __forceinline__ bf16x16 frag_b(const __bf16* base, int n,
                                                 int stride, int lane) {
  const __bf16* p = base + (size_t)(n + (lane & 15)) * stride + ((lane & 16) ? 8 : 0);
  bf16x8 lo = *(const bf16x8*)(p);
  bf16x8 hi = *(const bf16x8*)(p + 16);
  bf16x16 b;
#pragma unroll
  for (int i = 0; i < 8; ++i) { b[i] = lo[i]; b[i + 8] = hi[i]; }
  return b;
}

static __device__ __forceinline__ f32x8 wmma_bf16(bf16x16 a, bf16x16 b, f32x8 c) {
  return __builtin_amdgcn_wmma_f32_16x16x32_bf16(false, a, false, b, (short)0, c,
                                                 false, false);
}

static __device__ __forceinline__ float siluf(float y) {
  return y / (1.0f + __expf(-y));
}

// ---------------------------------------------------------------------------
// Row-wise f32 -> bf16 convert (one block per row; no div/mod)
// ---------------------------------------------------------------------------
__global__ __launch_bounds__(256)
void cvt_rows_kernel(const float* __restrict__ src, __bf16* __restrict__ dst, int cols) {
  const size_t base = (size_t)blockIdx.x * cols;
  for (int c = threadIdx.x; c < cols; c += 256)
    dst[base + c] = (__bf16)src[base + c];
}

// ---------------------------------------------------------------------------
// Tiled transpose + convert: src [K][N] f32 -> dst [rowoff+n][k] bf16
// (weights become [N][K] so GEMM B-tiles are plain row copies -> async DMA)
// ---------------------------------------------------------------------------
__global__ __launch_bounds__(256)
void cvt_transpose_kernel(const float* __restrict__ src, __bf16* __restrict__ dst,
                          int K, int N, int rowoff) {
  __shared__ float tile[32][33];
  const int tx = threadIdx.x & 31, ty = threadIdx.x >> 5;  // 32 x 8
  const int n0 = blockIdx.x * 32, k0 = blockIdx.y * 32;
#pragma unroll
  for (int i = 0; i < 4; ++i)
    tile[ty + i * 8][tx] = src[(size_t)(k0 + ty + i * 8) * N + n0 + tx];
  __syncthreads();
#pragma unroll
  for (int i = 0; i < 4; ++i)
    dst[(size_t)(rowoff + n0 + ty + i * 8) * K + k0 + tx] = (__bf16)tile[tx][ty + i * 8];
}

// ---------------------------------------------------------------------------
// bf16 GEMM: C[M,N] = A[M,K] * WT[N,K]^T. 128x256 workgroup tile, 8 waves,
// wave tile 64x64 (4x4 of 16x16 WMMA -> 16 wmma per 16 ds_load_b128).
// K-step 32, double-buffered LDS, async global->LDS staging when available.
// M % 128 == 0, N % 256 == 0, K % 32 == 0.
// ---------------------------------------------------------------------------
template <bool OUT_F32>
__global__ __launch_bounds__(256)
void gemm_bf16_kernel(const __bf16* __restrict__ A, const __bf16* __restrict__ WT,
                      void* __restrict__ Cout, int M, int N, int K) {
  __shared__ __bf16 sA[2][128 * 32];   // [m][k]
  __shared__ __bf16 sB[2][256 * 32];   // [n][k]

  const int tid  = threadIdx.x;
  const int lane = tid & 31;
  const int wave = tid >> 5;
  const int m0 = blockIdx.y * 128;
  const int n0 = blockIdx.x * 256;
  const int wm = (wave >> 2) * 64;
  const int wn = (wave & 3) * 64;

  const f32x8 vzero = {0.f, 0.f, 0.f, 0.f, 0.f, 0.f, 0.f, 0.f};
  f32x8 acc[4][4];
#pragma unroll
  for (int i = 0; i < 4; ++i)
#pragma unroll
    for (int j = 0; j < 4; ++j) acc[i][j] = vzero;

  // Staging: A 128x32 (2 thr/row, 16 elems), B 256x32 (1 thr/row, 32 elems).
  const int car = tid >> 1, cak = (tid & 1) * 16;

  auto stage = [&](int buf, int k0) {
    const __bf16* ga = A  + (size_t)(m0 + car) * K + k0 + cak;
    const __bf16* gb = WT + (size_t)(n0 + tid) * K + k0;
#ifdef HAVE_ASYNC_LDS
    async_copy16(ga,     &sA[buf][car * 32 + cak]);
    async_copy16(ga + 8, &sA[buf][car * 32 + cak + 8]);
#pragma unroll
    for (int i = 0; i < 4; ++i)
      async_copy16(gb + i * 8, &sB[buf][tid * 32 + i * 8]);
#else
    *(bf16x16*)&sA[buf][car * 32 + cak] = *(const bf16x16*)ga;
#pragma unroll
    for (int i = 0; i < 2; ++i)
      *(bf16x16*)&sB[buf][tid * 32 + i * 16] = *(const bf16x16*)(gb + i * 16);
#endif
  };

  stage(0, 0);
  const int nk = K >> 5;
  for (int ki = 0; ki < nk; ++ki) {
    const int cur = ki & 1;
#ifdef HAVE_ASYNC_LDS
    wait_async0();
#endif
    __syncthreads();
    if (ki + 1 < nk) stage(cur ^ 1, (ki + 1) << 5);

    bf16x16 af[4], bfr[4];
#pragma unroll
    for (int i = 0; i < 4; ++i) af[i] = frag_a(sA[cur], wm + i * 16, 32, lane);
#pragma unroll
    for (int j = 0; j < 4; ++j) bfr[j] = frag_b(sB[cur], wn + j * 16, 32, lane);
#pragma unroll
    for (int i = 0; i < 4; ++i)
#pragma unroll
      for (int j = 0; j < 4; ++j) acc[i][j] = wmma_bf16(af[i], bfr[j], acc[i][j]);
  }

  const int rl = lane & 15, rh = (lane >> 4) * 8;
#pragma unroll
  for (int i = 0; i < 4; ++i)
#pragma unroll
    for (int j = 0; j < 4; ++j) {
      const int mrow = m0 + wm + i * 16 + rh;
      const int ncol = n0 + wn + j * 16 + rl;
#pragma unroll
      for (int g = 0; g < 8; ++g) {
        const size_t idx = (size_t)(mrow + g) * N + ncol;
        if constexpr (OUT_F32) ((float*)Cout)[idx] = acc[i][j][g];
        else                   ((__bf16*)Cout)[idx] = (__bf16)acc[i][j][g];
      }
    }
}

// ---------------------------------------------------------------------------
// Causal depthwise conv (KW=4) + SiLU on fused qkv pre-activations,
// then per-head L2 norm of q (scaled 1/sqrt(D)) and k. One block per (b,t,h).
// ---------------------------------------------------------------------------
__global__ __launch_bounds__(128)
void conv_silu_norm_kernel(const __bf16* __restrict__ pre,
                           const float* __restrict__ qcw,
                           const float* __restrict__ kcw,
                           const float* __restrict__ vcw,
                           __bf16* __restrict__ qb, __bf16* __restrict__ kb,
                           __bf16* __restrict__ vb) {
  const int blk = blockIdx.x;
  const int h = blk % HH;
  const int t = (blk / HH) % LL;
  const int b = blk / (HH * LL);
  const int d = threadIdx.x;
  const int c = h * DD + d;
  const int lane = threadIdx.x & 31, wave = threadIdx.x >> 5;

  float aq = 0.f, akk = 0.f, av = 0.f;
#pragma unroll
  for (int j = 0; j < KWW; ++j) {
    const int tt = t - (KWW - 1) + j;
    if (tt >= 0) {
      const __bf16* p = pre + ((size_t)b * LL + tt) * (3 * HIDD);
      aq  += (float)p[c]            * qcw[c * KWW + j];
      akk += (float)p[HIDD + c]     * kcw[c * KWW + j];
      av  += (float)p[2 * HIDD + c] * vcw[c * KWW + j];
    }
  }
  const float q = siluf(aq), k = siluf(akk), v = siluf(av);

  __shared__ float redq[4], redk[4];
  float sq = q * q, sk = k * k;
#pragma unroll
  for (int off = 16; off > 0; off >>= 1) {
    sq += __shfl_down(sq, off, 32);
    sk += __shfl_down(sk, off, 32);
  }
  if (lane == 0) { redq[wave] = sq; redk[wave] = sk; }
  __syncthreads();
  const float nq = redq[0] + redq[1] + redq[2] + redq[3];
  const float nk = redk[0] + redk[1] + redk[2] + redk[3];

  const size_t o = ((size_t)b * LL + t) * HIDD + c;
  qb[o] = (__bf16)(q * rsqrtf(nq) * 0.08838834764831845f);  // * 1/sqrt(D)
  kb[o] = (__bf16)(k * rsqrtf(nk));
  vb[o] = (__bf16)v;
}

// ---------------------------------------------------------------------------
// beta = sigmoid(x @ b_w) ; b_w is [2048,16]. One block per row.
// ---------------------------------------------------------------------------
__global__ __launch_bounds__(256)
void beta_kernel(const float* __restrict__ x, const float* __restrict__ bw,
                 float* __restrict__ betab) {
  const int row = blockIdx.x;
  const int tid = threadIdx.x, lane = tid & 31, wave = tid >> 5;
  float xv[8];
#pragma unroll
  for (int i = 0; i < 8; ++i) xv[i] = x[(size_t)row * HIDD + tid + i * 256];
  float p[HH];
#pragma unroll
  for (int hh = 0; hh < HH; ++hh) p[hh] = 0.f;
#pragma unroll
  for (int i = 0; i < 8; ++i) {
    const float* w = bw + (size_t)(tid + i * 256) * HH;
#pragma unroll
    for (int hh = 0; hh < HH; ++hh) p[hh] += xv[i] * w[hh];
  }
  __shared__ float red[8 * HH];
#pragma unroll
  for (int hh = 0; hh < HH; ++hh) {
    float s = p[hh];
#pragma unroll
    for (int off = 16; off > 0; off >>= 1) s += __shfl_down(s, off, 32);
    if (lane == 0) red[wave * HH + hh] = s;
  }
  __syncthreads();
  if (tid < HH) {
    float s = 0.f;
#pragma unroll
    for (int w = 0; w < 8; ++w) s += red[w * HH + tid];
    betab[(size_t)row * HH + tid] = 1.f / (1.f + __expf(-s));
  }
}

// ---------------------------------------------------------------------------
// Chunked delta rule + fused RMS norm. One block (8 waves) per (b,h,n).
// S (last_state) is identically zero in the reference and never updated,
// so QSt/WSt/Wm vanish: O = (q kT (.) tril) @ (T @ v), T = inv(A) diag(beta),
// A = I + tril(diag(beta) k kT, -1). Output: rmsnorm(O)*o_norm_w as bf16.
// ---------------------------------------------------------------------------
__global__ __launch_bounds__(256)
void chunk_delta_kernel(const __bf16* __restrict__ qb, const __bf16* __restrict__ kb,
                        const __bf16* __restrict__ vb, const float* __restrict__ betab,
                        const float* __restrict__ onw, __bf16* __restrict__ onorm) {
  const int blk = blockIdx.x;
  const int n = blk % NCHUNK;
  const int h = (blk / NCHUNK) % HH;
  const int b = blk / (NCHUNK * HH);
  const int tid = threadIdx.x, lane = tid & 31, wave = tid >> 5;

  __shared__ __bf16 sK[CC * DD];      // k rows [c][d]
  __shared__ __bf16 sQUT[CC * DD];    // phase A: q rows [c][d]; phase B: U^T [d][c]
  __shared__ __bf16 sVT[DD * CC];     // v^T [d][c]
  __shared__ float  sA[CC * CC];      // A = I + tril(beta*KKt, -1)
  __shared__ float  sX[CC * CC];      // Tinv columns
  __shared__ __bf16 sTb[CC * CC];     // T bf16
  __shared__ __bf16 sMb[CC * CC];     // masked QKt bf16
  __shared__ float  sBeta[CC];
  __shared__ float  sRms[CC];         // per-row sum of squares of O
  __shared__ float  sOnw[DD];         // o_norm_w staged

  const size_t base = ((size_t)b * LL + (size_t)n * CC) * HIDD + (size_t)h * DD;
  const int r = tid >> 2, dq = (tid & 3) * 32;   // 64 rows, 32 elems/thread
  {
    const __bf16* gq = qb + base + (size_t)r * HIDD + dq;
    const __bf16* gk = kb + base + (size_t)r * HIDD + dq;
    const __bf16* gv = vb + base + (size_t)r * HIDD + dq;
#ifdef HAVE_ASYNC_LDS
#pragma unroll
    for (int i = 0; i < 4; ++i) {
      async_copy16(gq + i * 8, sQUT + r * DD + dq + i * 8);
      async_copy16(gk + i * 8, sK   + r * DD + dq + i * 8);
    }
#else
    *(bf16x16*)(sQUT + r * DD + dq)      = *(const bf16x16*)(gq);
    *(bf16x16*)(sQUT + r * DD + dq + 16) = *(const bf16x16*)(gq + 16);
    *(bf16x16*)(sK + r * DD + dq)        = *(const bf16x16*)(gk);
    *(bf16x16*)(sK + r * DD + dq + 16)   = *(const bf16x16*)(gk + 16);
#endif
    bf16x16 v0 = *(const bf16x16*)(gv);
    bf16x16 v1 = *(const bf16x16*)(gv + 16);
#pragma unroll
    for (int i = 0; i < 16; ++i) {
      sVT[(size_t)(dq + i) * CC + r]      = v0[i];
      sVT[(size_t)(dq + 16 + i) * CC + r] = v1[i];
    }
    if (tid < CC) {
      sBeta[tid] = betab[((size_t)b * LL + (size_t)n * CC + tid) * HH + h];
      sRms[tid] = 0.f;
    }
    if (tid < DD) sOnw[tid] = onw[tid];
  }
#ifdef HAVE_ASYNC_LDS
  wait_async0();
#endif
  __syncthreads();

  const int rl = lane & 15, rh = (lane >> 4) * 8;
  const f32x8 vzero = {0.f, 0.f, 0.f, 0.f, 0.f, 0.f, 0.f, 0.f};

  // ---- KKt -> A  (16 tiles, 2 per wave) ----
#pragma unroll
  for (int it = 0; it < 2; ++it) {
    const int t = wave * 2 + it;
    const int ti = (t >> 2) * 16, tj = (t & 3) * 16;
    f32x8 acc = vzero;
#pragma unroll
    for (int kk = 0; kk < DD; kk += 32)
      acc = wmma_bf16(frag_a(sK + kk, ti, DD, lane), frag_b(sK + kk, tj, DD, lane), acc);
    // One 32B vector load of the 8 consecutive beta rows this lane touches:
    // cannot be sunk into per-element predication by the compiler.
    const f32x8 bvec = *(const f32x8*)&sBeta[ti + rh];
#pragma unroll
    for (int g = 0; g < 8; ++g) {
      const int rr = ti + rh + g, cc = tj + rl;
      sA[rr * CC + cc] = (rr > cc) ? acc[g] * bvec[g] : (rr == cc ? 1.f : 0.f);
    }
  }
  __syncthreads();

  // ---- waves 4..7: masked QKt ; waves 0..1: triangular inverse + T ----
  if (wave >= 4) {
#pragma unroll
    for (int it = 0; it < 4; ++it) {
      const int t = (wave - 4) * 4 + it;
      const int ti = (t >> 2) * 16, tj = (t & 3) * 16;
      f32x8 acc = vzero;
#pragma unroll
      for (int kk = 0; kk < DD; kk += 32)
        acc = wmma_bf16(frag_a(sQUT + kk, ti, DD, lane), frag_b(sK + kk, tj, DD, lane), acc);
#pragma unroll
      for (int g = 0; g < 8; ++g) {
        const int rr = ti + rh + g, cc = tj + rl;
        sMb[rr * CC + cc] = (__bf16)((rr >= cc) ? acc[g] : 0.f);
      }
    }
  } else if (tid < CC) {
    const int cj = tid;   // columns are independent -> no intra-phase sync
    sX[0 * CC + cj] = (cj == 0) ? 1.f : 0.f;
    for (int rr2 = 1; rr2 < CC; ++rr2) {
      float accum = 0.f;
      for (int j = 0; j < rr2; ++j) accum += sA[rr2 * CC + j] * sX[j * CC + cj];
      sX[rr2 * CC + cj] = ((rr2 == cj) ? 1.f : 0.f) - accum;
    }
    const float bcj = sBeta[cj];
    for (int rr2 = 0; rr2 < CC; ++rr2)
      sTb[rr2 * CC + cj] = (__bf16)(sX[rr2 * CC + cj] * bcj);
  }
  __syncthreads();

  // ---- U = T @ v (32 tiles, 4/wave); store U^T into sQUT ----
#pragma unroll
  for (int it = 0; it < 4; ++it) {
    const int t = wave * 4 + it;
    const int mi = (t >> 3) * 16, nj = (t & 7) * 16;
    f32x8 acc = vzero;
#pragma unroll
    for (int kk = 0; kk < CC; kk += 32)
      acc = wmma_bf16(frag_a(sTb + kk, mi, CC, lane), frag_b(sVT + kk, nj, CC, lane), acc);
#pragma unroll
    for (int g = 0; g < 8; ++g)
      sQUT[(size_t)(nj + rl) * CC + mi + rh + g] = (__bf16)acc[g];
  }
  __syncthreads();

  // ---- O = (QKt (.) mask) @ U (32 tiles, 4/wave) + row sumsq via ds_add ----
  f32x8 oacc[4];
#pragma unroll
  for (int it = 0; it < 4; ++it) {
    const int t = wave * 4 + it;
    const int mi = (t >> 3) * 16, nj = (t & 7) * 16;
    f32x8 acc = vzero;
#pragma unroll
    for (int kk = 0; kk < CC; kk += 32)
      acc = wmma_bf16(frag_a(sMb + kk, mi, CC, lane), frag_b(sQUT + kk, nj, CC, lane), acc);
    oacc[it] = acc;
    // Row partial sums over this 16-column slice (lanes 0..15 / 16..31 are
    // the two row groups; xor masks < 16 reduce within each group).
#pragma unroll
    for (int g = 0; g < 8; ++g) {
      float p = acc[g] * acc[g];
#pragma unroll
      for (int m = 8; m >= 1; m >>= 1) p += __shfl_xor(p, m, 32);
      if ((lane & 15) == 0) atomicAdd(&sRms[mi + rh + g], p);
    }
  }
  __syncthreads();

  // ---- scale by rsqrt(mean+eps) * o_norm_w, emit bf16 for final GEMM ----
#pragma unroll
  for (int it = 0; it < 4; ++it) {
    const int t = wave * 4 + it;
    const int mi = (t >> 3) * 16, nj = (t & 7) * 16;
    const int row = mi + rh, col = nj + rl;
    const float w = sOnw[col];
    const f32x8 rvec = *(const f32x8*)&sRms[row];   // 8 consecutive rows, 32B
#pragma unroll
    for (int g = 0; g < 8; ++g) {
      const float rmsf = rsqrtf(rvec[g] * (1.0f / DD) + EPSF);
      onorm[((size_t)b * LL + (size_t)n * CC + row + g) * HIDD + (size_t)h * DD + col] =
          (__bf16)(oacc[it][g] * rmsf * w);
    }
  }
}

// ---------------------------------------------------------------------------
// Host-side launch
// ---------------------------------------------------------------------------
extern "C" void kernel_launch(void* const* d_in, const int* in_sizes, int n_in,
                              void* d_out, int out_size, void* d_ws, size_t ws_size,
                              hipStream_t stream) {
  const float* x   = (const float*)d_in[0];
  const float* lst = (const float*)d_in[1];
  const float* qw  = (const float*)d_in[2];
  const float* kw  = (const float*)d_in[3];
  const float* vw  = (const float*)d_in[4];
  const float* bw  = (const float*)d_in[5];
  const float* qcw = (const float*)d_in[6];
  const float* kcw = (const float*)d_in[7];
  const float* vcw = (const float*)d_in[8];
  const float* onw = (const float*)d_in[9];
  const float* ow  = (const float*)d_in[10];
  float* out = (float*)d_out;

  char* ws = (char*)d_ws;
  const size_t SZ_XB   = (size_t)BB * LL * HIDD * 2;
  const size_t SZ_WQKV = (size_t)HIDD * 3 * HIDD * 2;
  const size_t SZ_WO   = (size_t)HIDD * HIDD * 2;
  const size_t SZ_PRE  = (size_t)BB * LL * 3 * HIDD * 2;
  const size_t SZ_QKVB = (size_t)BB * LL * HIDD * 2;
  const size_t SZ_BETA = (size_t)BB * LL * HH * 4;
  __bf16* xb     = (__bf16*)(ws);
  __bf16* wqkvT  = (__bf16*)(ws + SZ_XB);                    // [3N][K] bf16
  __bf16* woT    = (__bf16*)(ws + SZ_XB + SZ_WQKV);          // [N][K] bf16
  __bf16* pre    = (__bf16*)(ws + SZ_XB + SZ_WQKV + SZ_WO);
  __bf16* qb     = (__bf16*)(ws + SZ_XB + SZ_WQKV + SZ_WO + SZ_PRE);
  __bf16* kb     = (__bf16*)((char*)qb + SZ_QKVB);
  __bf16* vb     = (__bf16*)((char*)kb + SZ_QKVB);
  float*  betab  = (float*)((char*)vb + SZ_QKVB);
  __bf16* onorm  = (__bf16*)((char*)betab + SZ_BETA);

  // Convert x (row copy) and weights (transpose to [N][K], q|k|v packed)
  cvt_rows_kernel<<<BB * LL, 256, 0, stream>>>(x, xb, HIDD);
  {
    dim3 tg(HIDD / 32, HIDD / 32);
    cvt_transpose_kernel<<<tg, 256, 0, stream>>>(qw, wqkvT, HIDD, HIDD, 0);
    cvt_transpose_kernel<<<tg, 256, 0, stream>>>(kw, wqkvT, HIDD, HIDD, HIDD);
    cvt_transpose_kernel<<<tg, 256, 0, stream>>>(vw, wqkvT, HIDD, HIDD, 2 * HIDD);
    cvt_transpose_kernel<<<tg, 256, 0, stream>>>(ow, woT, HIDD, HIDD, 0);
  }

  // Fused qkv GEMM: [8192 x 2048] x [2048 x 6144] -> bf16 pre-activations
  {
    dim3 grid(3 * HIDD / 256, BB * LL / 128);
    gemm_bf16_kernel<false><<<grid, 256, 0, stream>>>(xb, wqkvT, pre,
                                                      BB * LL, 3 * HIDD, HIDD);
  }

  conv_silu_norm_kernel<<<BB * LL * HH, DD, 0, stream>>>(pre, qcw, kcw, vcw, qb, kb, vb);
  beta_kernel<<<BB * LL, 256, 0, stream>>>(x, bw, betab);
  chunk_delta_kernel<<<BB * HH * NCHUNK, 256, 0, stream>>>(qb, kb, vb, betab, onw, onorm);

  // Output GEMM -> fp32 d_out
  {
    dim3 grid(HIDD / 256, BB * LL / 128);
    gemm_bf16_kernel<true><<<grid, 256, 0, stream>>>(onorm, woT, out,
                                                     BB * LL, HIDD, HIDD);
  }

  // Tuple tail: last_state passthrough (unchanged by the reference)
  (void)hipMemcpyAsync(out + (size_t)BB * LL * HIDD, lst,
                       (size_t)BB * 1 * HH * DD * DD * sizeof(float),
                       hipMemcpyDeviceToDevice, stream);
}
